// M_RNN_57501022159300
// MI455X (gfx1250) — matrix operand order repaired and checked by reading, not compiled, not run on hardware
//
#include <hip/hip_runtime.h>
#include <math.h>

// ---------------------------------------------------------------------------
// MI455X (gfx1250, wave32) implementation.
//
// Pipeline:
//   1) k_prep_swzB : swizzle w_net (512x256) and w_cls1 (256x256-as-B) into
//      CDNA5 WMMA B-fragment layout (bf16).
//   2) k_wcomb     : W_comb[h][v] = sum_e w_cls1[e][h]*emb[v][e] via WMMA,
//      stored directly in swizzled-B layout (16 MB, L2-resident).
//   3) k_rnn       : single workgroup (8 waves), w_net bf16 fragments held
//      resident in VGPRs (256 VGPRs/lane), per-step WMMA + LN + erf-GELU.
//   4) k_gemm      : logits = states @ W_comb + bias. 67 GFLOP but the
//      524 MB fp32 output store is the roofline (~22.5 us @ 23.3 TB/s),
//      so bf16 WMMA with fp32 accum + coalesced float4 stores.
// ---------------------------------------------------------------------------

typedef __attribute__((ext_vector_type(16))) __bf16 v16bf;
typedef __attribute__((ext_vector_type(8)))  float  v8f;
typedef unsigned short u16;
typedef unsigned int   u32;

#define B_   8
#define S_   512
#define E_   256
#define H_   256
#define V_   32000

// Native f32 -> bf16 (RNE) convert; backend emits hardware cvt.
__device__ __forceinline__ u16 f2bf(float f) {
  __bf16 h = (__bf16)f;
  return __builtin_bit_cast(u16, h);
}

// Convert 8 consecutive floats to 8 packed bf16 (one uint4).
__device__ __forceinline__ uint4 cvt8(const float* __restrict__ p) {
  float4 x = *(const float4*)p;
  float4 y = *(const float4*)(p + 4);
  uint4 r;
  r.x = (u32)f2bf(x.x) | ((u32)f2bf(x.y) << 16);
  r.y = (u32)f2bf(x.z) | ((u32)f2bf(x.w) << 16);
  r.z = (u32)f2bf(y.x) | ((u32)f2bf(y.y) << 16);
  r.w = (u32)f2bf(y.z) | ((u32)f2bf(y.w) << 16);
  return r;
}

__device__ __forceinline__ v8f wmma_bf(v16bf a, v16bf b, v8f c) {
  // D = A(16x32 bf16) x B(32x16 bf16) + C(16x16 f32)
  return __builtin_amdgcn_wmma_f32_16x16x32_bf16(false, a, false, b,
                                                 (short)0, c, false, false);
}

// A-operand (16-bit, 16x32) element->K map (ISA 7.12.2):
//   K = (e%8) + 16*(e/8) + 8*(lane/16)
// => a 16-element fragment group (fixed lane, chunk) is two contiguous
//    8-element K runs at k0 and k0+16, with k0 = c*32 + 8*(lane/16).
__device__ __forceinline__ int kkA(int e, int l) {
  return (e & 7) + 16 * (e >> 3) + 8 * (l >> 4);
}
// B-operand (16-bit, 32x16): K = (e%8) + 16*(e/8) + 8*(lane/16), N = lane%16
__device__ __forceinline__ int kkB(int e, int l) { return (e & 7) + 16 * (e >> 3) + 8 * (l >> 4); }

// ---------------------------------------------------------------------------
// Swizzle a row-major f32 matrix into bf16 B-fragment layout.
// dst flat index = [ntile][c][lane][e];  B[k][n] = src[n*strideN + k*strideK]
// ---------------------------------------------------------------------------
__global__ __launch_bounds__(256) void k_prep_swzB(
    const float* __restrict__ src, u16* __restrict__ dst,
    int kchunks, int strideN, int strideK, int total) {
  int idx = blockIdx.x * 256 + threadIdx.x;
  if (idx >= total) return;
  int e  = idx & 15;
  int l  = (idx >> 4) & 31;
  int c  = (idx >> 9) % kchunks;
  int nt = idx / (kchunks << 9);
  int n  = nt * 16 + (l & 15);
  int k  = c * 32 + kkB(e, l);
  dst[idx] = f2bf(src[(long)n * strideN + (long)k * strideK]);
}

// ---------------------------------------------------------------------------
// W_comb builder: P = emb(32000x256) @ w_cls1(256x256)  [K=E], then store
// P[v][h] transposed into swizzled-B layout (K=h, N=v).
// grid 500 blocks x 128 threads; wave w owns v-tile bx*4+w.
// ---------------------------------------------------------------------------
__global__ __launch_bounds__(128) void k_wcomb(
    const float* __restrict__ emb, const u16* __restrict__ wcls1_swz,
    u16* __restrict__ wcomb_swz) {
  __shared__ __align__(32) u16 As[4 * 8 * 32 * 16];  // 4 v-tiles, 32 KB
  int tid = threadIdx.x, lane = tid & 31, wv = tid >> 5;
  int bx = blockIdx.x;

  // Stage A (emb rows, f32 -> bf16), vectorized: 1024 groups of 16 elems.
#pragma unroll
  for (int i = 0; i < 8; ++i) {
    int g = i * 128 + tid;
    int l = g & 31, c = (g >> 5) & 7, tile = g >> 8;
    int v  = bx * 64 + tile * 16 + (l & 15);
    int k0 = c * 32 + 8 * (l >> 4);
    const float* src = &emb[(long)v * E_ + k0];
    *(uint4*)&As[g * 16]     = cvt8(src);
    *(uint4*)&As[g * 16 + 8] = cvt8(src + 16);
  }
  __syncthreads();

  v16bf a[8];
#pragma unroll
  for (int c = 0; c < 8; ++c)
    a[c] = *(const v16bf*)&As[((wv * 8 + c) * 32 + lane) * 16];

  int vtile = bx * 4 + wv;
  for (int n = 0; n < 16; ++n) {           // 16 h-tiles
    v8f acc = {};
#pragma unroll
    for (int c = 0; c < 8; ++c) {
      v16bf b = *(const v16bf*)&wcls1_swz[(((long)n * 8 + c) * 32 + lane) * 16];
      acc = wmma_bf(a[c], b, acc);
    }
    // D: lane holds column h = n*16 + lane%16, rows v = r + 8*(lane/16)
    int h  = n * 16 + (lane & 15);
    int cp = h >> 5, kp = h & 31, hi = (kp >> 3) & 1, ep = (kp & 7) + 8 * (kp >> 4);
#pragma unroll
    for (int r = 0; r < 8; ++r) {
      int vloc  = r + 8 * (lane >> 4);
      int lanep = vloc + 16 * hi;
      wcomb_swz[(((long)vtile * 8 + cp) * 32 + lanep) * 16 + ep] = f2bf(acc[r]);
    }
  }
}

// ---------------------------------------------------------------------------
// Recurrent scan: one workgroup, 8 waves. Each wave owns 2 h-tiles (32 cols)
// and keeps its 32 B-fragments of w_net resident in VGPRs (32 KB/wave).
// Per step: stage A=[emb(x_t); h] (16x512 bf16, rows 8-15 zero) in LDS,
// 32 WMMA per wave, LayerNorm (shuffle reduce) + exact GELU, h -> bf16.
// ---------------------------------------------------------------------------
__global__ __launch_bounds__(256) void k_rnn(
    const int* __restrict__ words, const float* __restrict__ emb,
    const u16* __restrict__ wnet_swz, const float* __restrict__ gamma,
    const float* __restrict__ beta, const float* __restrict__ h0,
    u16* __restrict__ states) {
  __shared__ __align__(32) u16  As[16 * 32 * 16];  // 16 K-chunks, 16 KB
  __shared__ __align__(32) u16  hls[B_ * H_];      // h state, bf16
  __shared__ float zls[B_ * H_];                   // pre-LN activations
  __shared__ float gls[H_], bls[H_];
  __shared__ int   wrow[B_];

  int tid = threadIdx.x, lane = tid & 31, wv = tid >> 5;
  gls[tid] = gamma[tid];
  bls[tid] = beta[tid];
  u16 h0b = f2bf(h0[tid]);
  for (int b = 0; b < B_; ++b) hls[b * H_ + tid] = h0b;

  // Resident w_net B-fragments: wave wv covers n-tiles 2*wv, 2*wv+1.
  v16bf Bf[2][16];
#pragma unroll
  for (int nt = 0; nt < 2; ++nt)
#pragma unroll
    for (int c = 0; c < 16; ++c)
      Bf[nt][c] = *(const v16bf*)&wnet_swz[(((wv * 2 + nt) * 16 + c) * 32 + lane) * 16];

  for (int t = 0; t < S_; ++t) {
    if (tid < B_) wrow[tid] = words[tid * S_ + t];
    __syncthreads();

    // Stage A: 512 groups of 16 elems; group g covers (chunk c, lane l).
    //   rows m<8: k<256 -> emb(x_t) (cvt), k>=256 -> h (bf16 copy); m>=8: 0.
#pragma unroll
    for (int i = 0; i < 2; ++i) {
      int g = i * 256 + tid;
      int l = g & 31, c = g >> 5;   // c in [0,16)
      int m = l & 15, hi = l >> 4;
      uint4 lo, hi4;
      if (m >= B_) {
        lo = make_uint4(0, 0, 0, 0);
        hi4 = lo;
      } else if (c < 8) {           // x_t part: k0 = c*32 + 8*hi < 256
        const float* src = &emb[(long)wrow[m] * E_ + c * 32 + 8 * hi];
        lo  = cvt8(src);
        hi4 = cvt8(src + 16);
      } else {                      // h part: (k-256) = (c-8)*32 + 8*hi
        int base = m * H_ + (c - 8) * 32 + 8 * hi;
        lo  = *(const uint4*)&hls[base];
        hi4 = *(const uint4*)&hls[base + 16];
      }
      *(uint4*)&As[g * 16]     = lo;
      *(uint4*)&As[g * 16 + 8] = hi4;
    }
    __syncthreads();

    v8f ac0 = {}, ac1 = {};
#pragma unroll
    for (int c = 0; c < 16; ++c) {
      v16bf a = *(const v16bf*)&As[(c * 32 + lane) * 16];
      ac0 = wmma_bf(a, Bf[0][c], ac0);
      ac1 = wmma_bf(a, Bf[1][c], ac1);
    }
    // D rows 0-7 (batch) live in lanes 0-15, VGPRs r=0..7
    if (lane < 16) {
#pragma unroll
      for (int r = 0; r < 8; ++r) {
        zls[r * H_ + wv * 32 + lane]      = ac0[r];
        zls[r * H_ + wv * 32 + 16 + lane] = ac1[r];
      }
    }
    __syncthreads();

    // LayerNorm + exact GELU: wave wv handles batch b = wv.
    {
      int b = wv;
      float s = 0.f, q = 0.f, zv[8];
#pragma unroll
      for (int i = 0; i < 8; ++i) {
        zv[i] = zls[b * H_ + lane + 32 * i];
        s += zv[i];
        q += zv[i] * zv[i];
      }
#pragma unroll
      for (int off = 16; off > 0; off >>= 1) {
        s += __shfl_xor(s, off, 32);
        q += __shfl_xor(q, off, 32);
      }
      float mu  = s * (1.f / 256.f);
      float var = q * (1.f / 256.f) - mu * mu;
      float rs  = rsqrtf(var + 1e-5f);
#pragma unroll
      for (int i = 0; i < 8; ++i) {
        int j = lane + 32 * i;
        float zn = (zv[i] - mu) * rs * gls[j] + bls[j];
        float g  = 0.5f * zn * (1.f + erff(zn * 0.70710678118654752f));
        u16 hb = f2bf(g);
        hls[b * H_ + j] = hb;
        states[((long)b * S_ + t) * H_ + j] = hb;
      }
    }
    __syncthreads();
  }
}

// ---------------------------------------------------------------------------
// logits = states(4096x256) @ W_comb(256x32000) + bias.  Store-bound.
// Block = 4 waves: 64 M-rows x 64 N-cols. Each wave: 1 n-tile x 4 m-tiles,
// so each coalesced 1 KB B-fragment load feeds 4 WMMAs. D restaged through
// LDS for contiguous float4 stores.
// ---------------------------------------------------------------------------
__global__ __launch_bounds__(128) void k_gemm(
    const u16* __restrict__ states, const u16* __restrict__ wcomb,
    const float* __restrict__ bias, float* __restrict__ out) {
  __shared__ __align__(32) u16  As[4 * 8 * 32 * 16];  // 4 m-tiles, 32 KB
  __shared__ __align__(16) float Ds[64 * 64];          // 16 KB
  int tid = threadIdx.x, lane = tid & 31, wv = tid >> 5;
  int bx = blockIdx.x, by = blockIdx.y;

  // Stage A (already bf16): 1024 groups of 16 elems, pure uint4 copies.
#pragma unroll
  for (int i = 0; i < 8; ++i) {
    int g = i * 128 + tid;
    int l = g & 31, c = (g >> 5) & 7, tile = g >> 8;
    int m = by * 64 + tile * 16 + (l & 15);
    const u16* src = &states[(long)m * H_ + c * 32 + 8 * (l >> 4)];
    *(uint4*)&As[g * 16]     = *(const uint4*)src;
    *(uint4*)&As[g * 16 + 8] = *(const uint4*)(src + 16);
  }
  __syncthreads();

  int ntile = bx * 4 + wv;
  v8f acc[4];
#pragma unroll
  for (int t4 = 0; t4 < 4; ++t4) acc[t4] = (v8f){};
#pragma unroll
  for (int c = 0; c < 8; ++c) {
    v16bf bfr = *(const v16bf*)&wcomb[(((long)ntile * 8 + c) * 32 + lane) * 16];
#pragma unroll
    for (int t4 = 0; t4 < 4; ++t4) {
      v16bf a = *(const v16bf*)&As[t4 * 4096 + (c * 32 + lane) * 16];
      acc[t4] = wmma_bf(a, bfr, acc[t4]);
    }
  }

  float bv = bias[ntile * 16 + (lane & 15)];
  int nl = wv * 16 + (lane & 15);
  int mb = 8 * (lane >> 4);
#pragma unroll
  for (int t4 = 0; t4 < 4; ++t4)
#pragma unroll
    for (int r = 0; r < 8; ++r)
      Ds[(t4 * 16 + mb + r) * 64 + nl] = acc[t4][r] + bv;
  __syncthreads();

  int row = tid >> 1, colg = (tid & 1) * 32;
  const float4* src = (const float4*)&Ds[row * 64 + colg];
  float4* dst = (float4*)&out[(long)(by * 64 + row) * V_ + bx * 64 + colg];
#pragma unroll
  for (int k4 = 0; k4 < 8; ++k4) dst[k4] = src[k4];
}

// ---------------------------------------------------------------------------
extern "C" void kernel_launch(void* const* d_in, const int* in_sizes, int n_in,
                              void* d_out, int out_size, void* d_ws,
                              size_t ws_size, hipStream_t stream) {
  const int*   words    = (const int*)d_in[0];
  const float* emb      = (const float*)d_in[1];
  const float* w_net    = (const float*)d_in[2];
  const float* ln_gamma = (const float*)d_in[3];
  const float* ln_beta  = (const float*)d_in[4];
  const float* h0       = (const float*)d_in[5];
  const float* w_cls1   = (const float*)d_in[6];
  const float* cls_bias = (const float*)d_in[7];
  float* out = (float*)d_out;

  char* ws = (char*)d_ws;
  u16* wnet_swz  = (u16*)(ws);              // 16*16*32*16  = 256 KB
  u16* wcls1_swz = (u16*)(ws + 262144);     // 16* 8*32*16  = 128 KB
  u16* wcomb_swz = (u16*)(ws + 393216);     // 2000*8*32*16 =  16 MB
  u16* states    = (u16*)(ws + 16777216);   // 8*512*256    =   2 MB

  // B[k][j] = w_net[j][k]  (N=256 -> 16 tiles, K=512 -> 16 chunks)
  k_prep_swzB<<<512, 256, 0, stream>>>(w_net, wnet_swz, 16, 512, 1, 131072);
  // B[e][h] = w_cls1[e][h] (N=256 -> 16 tiles, K=256 -> 8 chunks)
  k_prep_swzB<<<256, 256, 0, stream>>>(w_cls1, wcls1_swz, 8, 1, 256, 65536);
  // W_comb in swizzled-B layout
  k_wcomb<<<500, 128, 0, stream>>>(emb, wcls1_swz, wcomb_swz);
  // sequential RNN scan (single WGP, weights VGPR-resident)
  k_rnn<<<1, 256, 0, stream>>>(words, emb, wnet_swz, ln_gamma, ln_beta, h0,
                               states);
  // store-bound classifier GEMM
  k_gemm<<<dim3(500, 64), 128, 0, stream>>>(states, wcomb_swz, cls_bias, out);
}
